// SOT_71975061946520
// MI455X (gfx1250) — compile-verified, matrix operand order checked by reading, not run on hardware
//
#include <hip/hip_runtime.h>
#include <hip/hip_bf16.h>

#define N 8192
#define D 512
#define SLICES 16  // row slices for the column pass

typedef __attribute__((ext_vector_type(16))) __bf16 v16bf;
typedef __attribute__((ext_vector_type(8)))  __bf16 v8bf;
typedef __attribute__((ext_vector_type(8)))  float  v8f;

// ---- order-preserving float<->uint keys (for atomic max over possibly-negative floats)
__device__ __forceinline__ unsigned fkey(float f) {
  unsigned u = __float_as_uint(f);
  return (u >> 31) ? ~u : (u | 0x80000000u);
}
__device__ __forceinline__ float funkey(unsigned k) {
  return (k >> 31) ? __uint_as_float(k & 0x7fffffffu) : __uint_as_float(~k);
}

// ---------------------------------------------------------------- init
__global__ void k_init(float* lu, float* lv, unsigned* maxMbits, unsigned* maxLPkey) {
  int idx = blockIdx.x * 256 + threadIdx.x;
  if (idx < N) { lu[idx] = 0.f; lv[idx] = 0.f; }
  if (idx == 0) { *maxMbits = 0u; *maxLPkey = 0u; }
}

// ---------------------------------------------------------------- sq norms + bf16 hi/lo split
__global__ void __launch_bounds__(256) k_sqcvt(const float* __restrict__ X,
                                               unsigned short* __restrict__ Xhi_,
                                               unsigned short* __restrict__ Xlo_,
                                               float* __restrict__ sq) {
  __bf16* Xhi = (__bf16*)Xhi_;
  __bf16* Xlo = (__bf16*)Xlo_;
  int row = blockIdx.x, tid = threadIdx.x;
  float acc = 0.f;
  for (int c = tid; c < D; c += 256) {
    float x = X[(size_t)row * D + c];
    __bf16 h = (__bf16)x;           // RNE truncation to bf16
    float hf = (float)h;
    __bf16 l = (__bf16)(x - hf);    // residual
    Xhi[(size_t)row * D + c] = h;
    Xlo[(size_t)row * D + c] = l;
    acc += x * x;                   // exact-ish f32 row norm
  }
  __shared__ float sm[256];
  sm[tid] = acc; __syncthreads();
  for (int off = 128; off; off >>= 1) {
    if (tid < off) sm[tid] += sm[tid + off];
    __syncthreads();
  }
  if (tid == 0) sq[row] = sm[0];
}

// ---------------------------------------------------------------- WMMA tile loader (A-layout, 16x32 bf16)
__device__ __forceinline__ v16bf load_tile(const __bf16* __restrict__ base, int r0, int kk) {
  int lane = threadIdx.x & 31;
  const __bf16* p = base + (size_t)(r0 + (lane & 15)) * D + kk + ((lane >> 4) << 3);
  v8bf a = *(const v8bf*)(p);        // K chunk 0..7 (or 8..15)
  v8bf b = *(const v8bf*)(p + 16);   // K chunk 16..23 (or 24..31)
  return __builtin_shufflevector(a, b, 0,1,2,3,4,5,6,7,8,9,10,11,12,13,14,15);
}

// ---------------------------------------------------------------- Gram GEMM (bf16 hi/lo -> f32) + distance epilogue
// Each wave computes a 32x32 output region (2x2 WMMA tiles). X fits in L2, so direct global loads.
__global__ void __launch_bounds__(256) k_gemm(const unsigned short* __restrict__ Xhi_,
                                              const unsigned short* __restrict__ Xlo_,
                                              const float* __restrict__ sq,
                                              float* __restrict__ Mout,
                                              unsigned* __restrict__ maxMbits) {
  const __bf16* Xhi = (const __bf16*)Xhi_;
  const __bf16* Xlo = (const __bf16*)Xlo_;
  int wave = (blockIdx.x << 3) + (threadIdx.x >> 5);
  int rb = (wave >> 8) * 32;     // 256 regions per dim
  int cb = (wave & 255) * 32;

  v8f acc[2][2] = {};
  for (int kk = 0; kk < D; kk += 32) {
    v16bf ah0 = load_tile(Xhi, rb,      kk);
    v16bf ah1 = load_tile(Xhi, rb + 16, kk);
    v16bf bh0 = load_tile(Xhi, cb,      kk);
    v16bf bh1 = load_tile(Xhi, cb + 16, kk);
    // hi*hi
    acc[0][0] = __builtin_amdgcn_wmma_f32_16x16x32_bf16(false, ah0, false, bh0, (short)0, acc[0][0], false, false);
    acc[0][1] = __builtin_amdgcn_wmma_f32_16x16x32_bf16(false, ah0, false, bh1, (short)0, acc[0][1], false, false);
    acc[1][0] = __builtin_amdgcn_wmma_f32_16x16x32_bf16(false, ah1, false, bh0, (short)0, acc[1][0], false, false);
    acc[1][1] = __builtin_amdgcn_wmma_f32_16x16x32_bf16(false, ah1, false, bh1, (short)0, acc[1][1], false, false);
    // hi*lo
    v16bf bl0 = load_tile(Xlo, cb,      kk);
    v16bf bl1 = load_tile(Xlo, cb + 16, kk);
    acc[0][0] = __builtin_amdgcn_wmma_f32_16x16x32_bf16(false, ah0, false, bl0, (short)0, acc[0][0], false, false);
    acc[0][1] = __builtin_amdgcn_wmma_f32_16x16x32_bf16(false, ah0, false, bl1, (short)0, acc[0][1], false, false);
    acc[1][0] = __builtin_amdgcn_wmma_f32_16x16x32_bf16(false, ah1, false, bl0, (short)0, acc[1][0], false, false);
    acc[1][1] = __builtin_amdgcn_wmma_f32_16x16x32_bf16(false, ah1, false, bl1, (short)0, acc[1][1], false, false);
    // lo*hi
    v16bf al0 = load_tile(Xlo, rb,      kk);
    v16bf al1 = load_tile(Xlo, rb + 16, kk);
    acc[0][0] = __builtin_amdgcn_wmma_f32_16x16x32_bf16(false, al0, false, bh0, (short)0, acc[0][0], false, false);
    acc[0][1] = __builtin_amdgcn_wmma_f32_16x16x32_bf16(false, al0, false, bh1, (short)0, acc[0][1], false, false);
    acc[1][0] = __builtin_amdgcn_wmma_f32_16x16x32_bf16(false, al1, false, bh0, (short)0, acc[1][0], false, false);
    acc[1][1] = __builtin_amdgcn_wmma_f32_16x16x32_bf16(false, al1, false, bh1, (short)0, acc[1][1], false, false);
  }

  // epilogue: d2 = sq_r + sq_c - 2*G ; M = sqrt(max(d2,0)); fused global max(M)
  int lane = threadIdx.x & 31;
  int cn = lane & 15;
  int rAdd = (lane >> 4) << 3;   // C/D layout: lanes 16-31 hold M = v+8
  float lmax = 0.f;
  for (int ti = 0; ti < 2; ++ti)
    for (int tj = 0; tj < 2; ++tj)
      for (int v = 0; v < 8; ++v) {
        int r = rb + ti * 16 + rAdd + v;
        int c = cb + tj * 16 + cn;
        float g = acc[ti][tj][v];
        float d2 = sq[r] + sq[c] - 2.f * g;
        float m = sqrtf(fmaxf(d2, 0.f));
        Mout[(size_t)r * N + c] = m;
        lmax = fmaxf(lmax, m);
      }
  __shared__ unsigned smax;
  if (threadIdx.x == 0) smax = 0u;
  __syncthreads();
  atomicMax(&smax, __float_as_uint(lmax));   // non-negative floats: bits are monotone
  __syncthreads();
  if (threadIdx.x == 0) atomicMax(maxMbits, smax);
}

// ---------------------------------------------------------------- Sinkhorn row pass: log_u[i] = -LSE_j(log_K[i,j] + log_v[j])
__global__ void __launch_bounds__(256) k_rowpass(const float* __restrict__ M,
                                                 const float* __restrict__ lv,
                                                 float* __restrict__ lu,
                                                 const unsigned* __restrict__ maxMbits) {
  int i = blockIdx.x, tid = threadIdx.x;
  float scale = 10.f / __uint_as_float(*maxMbits);   // 1/OT_REG / maxM
  float m = -3.0e38f, s = 0.f;
  for (int j = tid; j < N; j += 256) {
    float lk = (j == i) ? -10000.f : -scale * M[(size_t)i * N + j];
    float a = lk + lv[j];
    if (a > m) { s = s * __expf(m - a) + 1.f; m = a; }
    else       { s += __expf(a - m); }
  }
  __shared__ float sm[256], ss[256];
  sm[tid] = m; ss[tid] = s; __syncthreads();
  for (int off = 128; off; off >>= 1) {
    if (tid < off) {
      float m2 = sm[tid + off], s2 = ss[tid + off];
      float mm = fmaxf(sm[tid], m2);
      ss[tid] = ss[tid] * __expf(sm[tid] - mm) + s2 * __expf(m2 - mm);
      sm[tid] = mm;
    }
    __syncthreads();
  }
  if (tid == 0) lu[i] = -(sm[0] + __logf(ss[0]));
}

// ---------------------------------------------------------------- Sinkhorn col pass (2D-blocked): partial LSE per 512-row slice
__global__ void __launch_bounds__(256) k_colpass(const float* __restrict__ M,
                                                 const float* __restrict__ lu,
                                                 float* __restrict__ pm,
                                                 float* __restrict__ ps,
                                                 const unsigned* __restrict__ maxMbits) {
  int j = blockIdx.x * 256 + threadIdx.x;
  int i0 = blockIdx.y * (N / SLICES);
  float scale = 10.f / __uint_as_float(*maxMbits);
  float m = -3.0e38f, s = 0.f;
#pragma unroll 4
  for (int i = i0; i < i0 + N / SLICES; ++i) {
    float lk = (i == j) ? -10000.f : -scale * M[(size_t)i * N + j];
    float a = lu[i] + lk;
    if (a > m) { s = s * __expf(m - a) + 1.f; m = a; }
    else       { s += __expf(a - m); }
  }
  pm[(size_t)blockIdx.y * N + j] = m;
  ps[(size_t)blockIdx.y * N + j] = s;
}

__global__ void k_colcombine(const float* __restrict__ pm, const float* __restrict__ ps,
                             float* __restrict__ lv) {
  int j = blockIdx.x * 256 + threadIdx.x;
  float m = -3.0e38f, s = 0.f;
#pragma unroll
  for (int t = 0; t < SLICES; ++t) {
    float m2 = pm[(size_t)t * N + j], s2 = ps[(size_t)t * N + j];
    float mm = fmaxf(m, m2);
    s = s * __expf(m - mm) + s2 * __expf(m2 - mm);
    m = mm;
  }
  lv[j] = -(m + __logf(s));
}

// ---------------------------------------------------------------- max over log_P (sign-safe keyed atomic max)
__global__ void __launch_bounds__(256) k_maxlogp(const float* __restrict__ M,
                                                 const float* __restrict__ lu,
                                                 const float* __restrict__ lv,
                                                 const unsigned* __restrict__ maxMbits,
                                                 unsigned* __restrict__ maxLPkey) {
  float scale = 10.f / __uint_as_float(*maxMbits);
  float lmax = -3.0e38f;
  size_t total = (size_t)N * N;
  size_t stride = (size_t)gridDim.x * blockDim.x;
  for (size_t idx = (size_t)blockIdx.x * blockDim.x + threadIdx.x; idx < total; idx += stride) {
    int i = (int)(idx >> 13);
    int j = (int)(idx & (N - 1));
    float lk = (i == j) ? -10000.f : -scale * M[idx];
    lmax = fmaxf(lmax, lu[i] + lk + lv[j]);
  }
  __shared__ unsigned sk;
  if (threadIdx.x == 0) sk = 0u;
  __syncthreads();
  atomicMax(&sk, fkey(lmax));
  __syncthreads();
  if (threadIdx.x == 0) atomicMax(maxLPkey, sk);
}

// ---------------------------------------------------------------- finalize: P = exp(logP - maxLogP), diag = 1 (in place)
__global__ void __launch_bounds__(256) k_final(float* __restrict__ M,
                                               const float* __restrict__ lu,
                                               const float* __restrict__ lv,
                                               const unsigned* __restrict__ maxMbits,
                                               const unsigned* __restrict__ maxLPkey) {
  float scale = 10.f / __uint_as_float(*maxMbits);
  float mlp = funkey(*maxLPkey);
  size_t total = (size_t)N * N;
  size_t stride = (size_t)gridDim.x * blockDim.x;
  for (size_t idx = (size_t)blockIdx.x * blockDim.x + threadIdx.x; idx < total; idx += stride) {
    int i = (int)(idx >> 13);
    int j = (int)(idx & (N - 1));
    float p;
    if (i == j) p = 1.f;
    else        p = __expf(lu[i] + (-scale * M[idx]) + lv[j] - mlp);
    M[idx] = p;
  }
}

// ---------------------------------------------------------------- launch
extern "C" void kernel_launch(void* const* d_in, const int* in_sizes, int n_in,
                              void* d_out, int out_size, void* d_ws, size_t ws_size,
                              hipStream_t stream) {
  const float* X = (const float*)d_in[0];
  float* out = (float*)d_out;   // holds M_unnorm through Sinkhorn, then final P

  char* w = (char*)d_ws;
  unsigned short* Xhi = (unsigned short*)w; w += (size_t)N * D * 2;   // 8 MB
  unsigned short* Xlo = (unsigned short*)w; w += (size_t)N * D * 2;   // 8 MB
  float* sq = (float*)w;  w += (size_t)N * 4;
  float* lu = (float*)w;  w += (size_t)N * 4;
  float* lv = (float*)w;  w += (size_t)N * 4;
  float* pm = (float*)w;  w += (size_t)SLICES * N * 4;                // 512 KB
  float* ps = (float*)w;  w += (size_t)SLICES * N * 4;                // 512 KB
  unsigned* maxMbits = (unsigned*)w; w += 64;
  unsigned* maxLPkey = (unsigned*)w;

  k_init<<<(N + 255) / 256, 256, 0, stream>>>(lu, lv, maxMbits, maxLPkey);
  k_sqcvt<<<N, 256, 0, stream>>>(X, Xhi, Xlo, sq);
  k_gemm<<<(N / 32) * (N / 32) / 8, 256, 0, stream>>>(Xhi, Xlo, sq, out, maxMbits);

  for (int it = 0; it < 10; ++it) {
    k_rowpass<<<N, 256, 0, stream>>>(out, lv, lu, maxMbits);
    k_colpass<<<dim3(N / 256, SLICES), 256, 0, stream>>>(out, lu, pm, ps, maxMbits);
    k_colcombine<<<N / 256, 256, 0, stream>>>(pm, ps, lv);
  }

  k_maxlogp<<<16384, 256, 0, stream>>>(out, lu, lv, maxMbits, maxLPkey);
  k_final<<<16384, 256, 0, stream>>>(out, lu, lv, maxMbits, maxLPkey);
}
